// Set2Vec_19868518711813
// MI455X (gfx1250) — matrix operand order repaired and checked by reading, not compiled, not run on hardware
//
#include <hip/hip_runtime.h>
#include <hip/hip_bf16.h>

typedef __attribute__((ext_vector_type(16))) _Float16 v16h;
typedef __attribute__((ext_vector_type(8)))  _Float16 v8h;
typedef __attribute__((ext_vector_type(4)))  _Float16 v4h;
typedef __attribute__((ext_vector_type(8)))  float    v8f;

#define N_NODES 100
#define MDIM    128
#define KCAT    256
#define BIG_NEG (-1.0e6f)

// ---------------------------------------------------------------------------
// WMMA fragment loaders (CDNA5 16-bit layouts, CDNA5 ISA §7.12.2)
// ---------------------------------------------------------------------------
// A 16x32 f16 from an f32 source (fused convert): lanes 0-15 row M=lane,
// K=0..7 / 16..23; lanes 16-31 same rows, K=8..15 / 24..31.
__device__ inline v16h load_frag_a_f32(const float* __restrict__ src, int ld, int lane) {
  const int r  = lane & 15;
  const int kh = (lane >> 4) << 3;            // 0 or 8
  const float* p = src + (size_t)r * ld + kh;
  v16h out;
#pragma unroll
  for (int v = 0; v < 8; ++v) out[v]     = (_Float16)p[v];        // K=kh+0..7
#pragma unroll
  for (int v = 0; v < 8; ++v) out[8 + v] = (_Float16)p[16 + v];   // K=kh+16..23
  return out;
}

// A 16x32 f16 from an f16 (LDS) tile, same layout.
__device__ inline v16h load_frag_a(const _Float16* __restrict__ base, int ld, int lane) {
  const int r  = lane & 15;
  const int kh = (lane >> 4) << 3;
  const _Float16* p = base + r * ld + kh;
  v16h out;
#pragma unroll
  for (int v = 0; v < 8; ++v) out[v]     = p[v];
#pragma unroll
  for (int v = 0; v < 8; ++v) out[8 + v] = p[16 + v];
  return out;
}

// B 32x16 f16: lane holds column n=lane&15; lanes 0-15 K=0..7 / 16..23, lanes 16-31
// K=8..15 / 24..31. colbase -> column-0 K-run; ld = per-column stride ([n][k] storage).
__device__ inline v16h load_frag_b(const _Float16* __restrict__ colbase, int ld, int lane) {
  const int n  = lane & 15;
  const int k0 = (lane >> 4) << 3;            // 0 or 8
  const _Float16* p = colbase + n * ld + k0;
  v16h out;
#pragma unroll
  for (int v = 0; v < 8; ++v) { out[v] = p[v]; out[8 + v] = p[16 + v]; }
  return out;
}

// ---------------------------------------------------------------------------
// Kernel 0: convert weights f32 -> f16 into workspace
// layout: [W_emb 128*256][W_ih 512*128][W_hh 512*128][memory B*N*128]
// ---------------------------------------------------------------------------
__global__ void cvt_weights(const float* __restrict__ wemb,
                            const float* __restrict__ wih,
                            const float* __restrict__ whh,
                            _Float16* __restrict__ dst) {
  int i = blockIdx.x * blockDim.x + threadIdx.x;
  if (i < 32768)            dst[i] = (_Float16)wemb[i];
  else if (i < 32768+65536) dst[i] = (_Float16)wih[i - 32768];
  else if (i < 163840)      dst[i] = (_Float16)whh[i - 98304];
}

// ---------------------------------------------------------------------------
// Kernel 1: memory = concat(hidden,input) @ W_emb^T + b_emb   (f16 out)
// 128-row x 128-col tile per block, 8 waves (16-row strip each), K=256.
// A fragments loaded straight from global f32 (rows are wave-exclusive ->
// no reuse to exploit; skip LDS and barriers entirely).
// ---------------------------------------------------------------------------
__global__ __launch_bounds__(256)
void embed_gemm(const float* __restrict__ hidden,
                const float* __restrict__ input,
                const float* __restrict__ b_emb,
                const _Float16* __restrict__ wembH,
                _Float16* __restrict__ memH) {
  const int lane = threadIdx.x & 31;
  const int wave = threadIdx.x >> 5;
  const int row0 = blockIdx.x * 128 + wave * 16;  // B*N = 1600*128 exactly

  v8f acc[8] = {};

#pragma unroll
  for (int kc = 0; kc < 8; ++kc) {
    const float* src = (kc < 4) ? hidden : input;
    const int kbase  = (kc < 4) ? kc * 32 : (kc - 4) * 32;
    v16h a = load_frag_a_f32(src + (size_t)row0 * MDIM + kbase, MDIM, lane);
#pragma unroll
    for (int ct = 0; ct < 8; ++ct) {
      v16h b = load_frag_b(wembH + (size_t)(ct * 16) * KCAT + kc * 32, KCAT, lane);
      acc[ct] = __builtin_amdgcn_wmma_f32_16x16x32_f16(
          false, a, false, b, (short)0, acc[ct], false, false);
    }
  }

  // D layout: VGPR r -> row r (+8 for lanes>=16); col = lane&15
  const int cl  = lane & 15;
  const int rhi = (lane >> 4) * 8;
#pragma unroll
  for (int ct = 0; ct < 8; ++ct) {
    const int col = ct * 16 + cl;
    const float bias = b_emb[col];
#pragma unroll
    for (int r = 0; r < 8; ++r)
      memH[(size_t)(row0 + rhi + r) * MDIM + col] = (_Float16)(acc[ct][r] + bias);
  }
}

// ---------------------------------------------------------------------------
// Kernel 2: persistent recurrent kernel. One block = 16 batch rows, 16 waves.
// All T steps in-kernel; weight WMMA fragments hoisted out of the serial loop;
// memory tiles stream from the L2-resident f16 buffer with 128-bit loads.
// ---------------------------------------------------------------------------
__global__ __launch_bounds__(512)
void s2v_recurrent(const _Float16* __restrict__ memH,
                   const _Float16* __restrict__ wihH,
                   const _Float16* __restrict__ whhH,
                   const float* __restrict__ b_ih,
                   const float* __restrict__ b_hh,
                   const unsigned char* __restrict__ mask,
                   const int* __restrict__ Tptr,
                   float* __restrict__ out) {
  __shared__ float    hS[16][132];            // q / hidden
  __shared__ float    cS[16][132];            // cell
  __shared__ float    xS[16][132];            // lstm_in (attention read)
  __shared__ _Float16 Xh[16][136];            // f16 staging for WMMA A
  __shared__ _Float16 Hh[16][136];
  __shared__ float    gS[16][516];            // gates; attn overlays (dead by then)
  float* aS = &gS[0][0];                      // attn[16][104]

  const int tid   = threadIdx.x;
  const int lane  = tid & 31;
  const int wave  = tid >> 5;
  const int bb    = blockIdx.x * 16 + wave;   // this wave's batch row
  const int T     = Tptr[0];

  for (int idx = tid; idx < 16 * 128; idx += 512) {
    const int r = idx >> 7, cc = idx & 127;
    hS[r][cc] = 0.f; cS[r][cc] = 0.f; xS[r][cc] = 0.f;
  }

  const _Float16* mrow = memH + (size_t)bb * N_NODES * MDIM;
  const unsigned char* mk = mask + (size_t)bb * N_NODES;

  // ---- hoist loop-invariants out of the serial T chain --------------------
  // Weight fragments: [half][kc][0]=W_ih, [half][kc][1]=W_hh  (128 VGPRs)
  v16h wfrag[2][4][2];
  float bs[2];
#pragma unroll
  for (int half = 0; half < 2; ++half) {
    const int ct = wave + half * 16;
#pragma unroll
    for (int kc = 0; kc < 4; ++kc) {
      wfrag[half][kc][0] = load_frag_b(wihH + (size_t)(ct * 16) * MDIM + kc * 32, MDIM, lane);
      wfrag[half][kc][1] = load_frag_b(whhH + (size_t)(ct * 16) * MDIM + kc * 32, MDIM, lane);
    }
    const int col = ct * 16 + (lane & 15);
    bs[half] = b_ih[col] + b_hh[col];
  }
  // Energy mask bias per lane-owned node
  float emask[4];
#pragma unroll
  for (int j = 0; j < 4; ++j) {
    const int n = lane + 32 * j;
    emask[j] = (n < N_NODES) ? (mk[n] ? 0.f : BIG_NEG) : 0.f;
  }
  __syncthreads();

  for (int t = 0; t < T; ++t) {
    // -- stage f16 operands ------------------------------------------------
    for (int idx = tid; idx < 16 * 128; idx += 512) {
      const int r = idx >> 7, cc = idx & 127;
      Xh[r][cc] = (_Float16)xS[r][cc];
      Hh[r][cc] = (_Float16)hS[r][cc];
    }
    __syncthreads();

    // -- gates = X@Wih^T + H@Whh^T + b  (WMMA, 2 col-tiles per wave) -------
    v16h ax[4], ah[4];
#pragma unroll
    for (int kc = 0; kc < 4; ++kc) {
      ax[kc] = load_frag_a(&Xh[0][kc * 32], 136, lane);
      ah[kc] = load_frag_a(&Hh[0][kc * 32], 136, lane);
    }
#pragma unroll
    for (int half = 0; half < 2; ++half) {
      const int ct = wave + half * 16;
      v8f acc = {};
#pragma unroll
      for (int kc = 0; kc < 4; ++kc) {
        acc = __builtin_amdgcn_wmma_f32_16x16x32_f16(
            false, ax[kc], false, wfrag[half][kc][0], (short)0, acc, false, false);
        acc = __builtin_amdgcn_wmma_f32_16x16x32_f16(
            false, ah[kc], false, wfrag[half][kc][1], (short)0, acc, false, false);
      }
      const int col = ct * 16 + (lane & 15);
      const int rhi = (lane >> 4) * 8;
#pragma unroll
      for (int r = 0; r < 8; ++r) gS[rhi + r][col] = acc[r] + bs[half];
    }
    __syncthreads();

    // -- LSTM cell elementwise (gate order i,f,g,o) ------------------------
    for (int idx = tid; idx < 16 * 128; idx += 512) {
      const int r = idx >> 7, cc = idx & 127;
      const float ig = gS[r][cc], fg = gS[r][128 + cc];
      const float gg = gS[r][256 + cc], og = gS[r][384 + cc];
      const float si = 1.f / (1.f + __expf(-ig));
      const float sf = 1.f / (1.f + __expf(-fg));
      const float so = 1.f / (1.f + __expf(-og));
      const float c  = sf * cS[r][cc] + si * tanhf(gg);
      cS[r][cc] = c;
      hS[r][cc] = so * tanhf(c);              // q
    }
    __syncthreads();

    // -- energies: e[n] = q . memory[n,:] + mask  (wave owns batch row) ----
    const float4* qp = (const float4*)&hS[wave][0];   // row start 16B-aligned
    float e[4];
#pragma unroll
    for (int j = 0; j < 4; ++j) {
      const int n = lane + 32 * j;
      float v = -1.0e30f;
      if (n < N_NODES) {
        const v8h* mp8 = (const v8h*)(mrow + (size_t)n * MDIM);
        float s = 0.f;
#pragma unroll
        for (int i = 0; i < 16; ++i) {
          const v8h m8 = mp8[i];
          const float4 q0 = qp[2 * i];
          const float4 q1 = qp[2 * i + 1];
          s += q0.x * (float)m8[0] + q0.y * (float)m8[1]
             + q0.z * (float)m8[2] + q0.w * (float)m8[3]
             + q1.x * (float)m8[4] + q1.y * (float)m8[5]
             + q1.z * (float)m8[6] + q1.w * (float)m8[7];
        }
        v = s + emask[j];
      }
      e[j] = v;
    }
    // wave32 softmax
    float mx = fmaxf(fmaxf(e[0], e[1]), fmaxf(e[2], e[3]));
#pragma unroll
    for (int off = 16; off > 0; off >>= 1) mx = fmaxf(mx, __shfl_xor(mx, off, 32));
    float sum = 0.f;
#pragma unroll
    for (int j = 0; j < 4; ++j) {
      e[j] = (lane + 32 * j < N_NODES) ? __expf(e[j] - mx) : 0.f;
      sum += e[j];
    }
#pragma unroll
    for (int off = 16; off > 0; off >>= 1) sum += __shfl_xor(sum, off, 32);
    const float inv = 1.f / sum;
#pragma unroll
    for (int j = 0; j < 4; ++j) {
      const int n = lane + 32 * j;
      if (n < N_NODES) aS[wave * 104 + n] = e[j] * inv;
    }
    __syncthreads();

    // -- read: x[k] = sum_n attn[n]*memory[n,k]; lane owns k=4*lane..+3 ----
    {
      float r0 = 0.f, r1 = 0.f, r2 = 0.f, r3 = 0.f;
      const _Float16* mcol = mrow + lane * 4;
      for (int n = 0; n < N_NODES; ++n) {
        const float a = aS[wave * 104 + n];
        const v4h m4 = *(const v4h*)(mcol + (size_t)n * MDIM);   // 8B-aligned
        r0 += a * (float)m4[0];
        r1 += a * (float)m4[1];
        r2 += a * (float)m4[2];
        r3 += a * (float)m4[3];
      }
      xS[wave][lane * 4 + 0] = r0;
      xS[wave][lane * 4 + 1] = r1;
      xS[wave][lane * 4 + 2] = r2;
      xS[wave][lane * 4 + 3] = r3;
    }
    __syncthreads();
  }

  // output: [q (128) | read (128)] per batch row
#pragma unroll
  for (int j = 0; j < 4; ++j) {
    const int k = lane + 32 * j;
    out[(size_t)bb * 256 + k]       = hS[wave][k];
    out[(size_t)bb * 256 + 128 + k] = xS[wave][k];
  }
}

// ---------------------------------------------------------------------------
extern "C" void kernel_launch(void* const* d_in, const int* in_sizes, int n_in,
                              void* d_out, int out_size, void* d_ws, size_t ws_size,
                              hipStream_t stream) {
  const float* hidden = (const float*)d_in[0];   // [B,N,128]
  const float* input  = (const float*)d_in[1];   // [B,N,128]
  const float* W_emb  = (const float*)d_in[2];   // [128,256]
  const float* b_emb  = (const float*)d_in[3];   // [128]
  const float* W_ih   = (const float*)d_in[4];   // [512,128]
  const float* b_ih   = (const float*)d_in[5];   // [512]
  const float* W_hh   = (const float*)d_in[6];   // [512,128]
  const float* b_hh   = (const float*)d_in[7];   // [512]
  const unsigned char* node_mask = (const unsigned char*)d_in[8]; // bool [B,N]
  const int* Tptr     = (const int*)d_in[9];     // scalar T (read on device)

  const int BN = in_sizes[8];                    // B*N
  const int B  = BN / N_NODES;

  _Float16* wembH = (_Float16*)d_ws;             // 128*256
  _Float16* wihH  = wembH + 128 * 256;           // 512*128
  _Float16* whhH  = wihH + 512 * 128;            // 512*128
  _Float16* memH  = whhH + 512 * 128;            // B*N*128 f16 (L2-resident)

  cvt_weights<<<(163840 + 255) / 256, 256, 0, stream>>>(W_emb, W_ih, W_hh, wembH);
  embed_gemm<<<BN / 128, 256, 0, stream>>>(hidden, input, b_emb, wembH, memH);
  s2v_recurrent<<<B / 16, 512, 0, stream>>>(memH, wihH, whhH, b_ih, b_hh,
                                            node_mask, Tptr, (float*)d_out);
}
